// Trigger_Recognition_77738908058180
// MI455X (gfx1250) — matrix operand order repaired and checked by reading, not compiled, not run on hardware
//
#include <hip/hip_runtime.h>
#include <hip/hip_bf16.h>
#include <math.h>

typedef __attribute__((ext_vector_type(16))) __bf16 v16bf;
typedef __attribute__((ext_vector_type(8)))  float  v8f;

#define T_STEPS 65536
#define HIDDEN  256
#define EMB     16
#define TMP     128
#define EVD     128
#define NEG     0.01f

// ---- workspace layout (bytes) ----
#define WS_Y      0                              // T*128 f32   = 33554432
#define WS_ARG    33554432                       // T*128 u8    = 8388608
#define WS_E      41943040                       // 128*128 f32 = 65536
#define WS_CM     42008576                       // 128*128 u8  = 16384
#define WS_START  42024960                       // 128 i32
#define WS_PIDX   42025472                       // T i32
#define WS_NIDX   42287616                       // T i32

__device__ __forceinline__ float lrelu(float x) { return x > 0.0f ? x : NEG * x; }

// ---------------- Kernel 0: E[j][k] = emb[j] @ W1[256: , k] ----------------
__global__ __launch_bounds__(256) void k_prepE(const float* __restrict__ emb,
                                               const float* __restrict__ W1,
                                               float* __restrict__ E) {
    int idx = blockIdx.x * 256 + threadIdx.x;
    if (idx >= 128 * 128) return;
    int j = idx >> 7, k = idx & 127;
    float s = 0.0f;
#pragma unroll
    for (int e = 0; e < EMB; ++e)
        s += emb[j * EMB + e] * W1[(HIDDEN + e) * TMP + k];
    E[idx] = s;
}

// ---------------- Kernel 1: Y = X @ W1[:256] + b1  (bf16 WMMA) ----------------
// block = 256 threads (8 waves), each block does 128 timesteps (wave = m-tile)
__global__ __launch_bounds__(256) void k_gemm1(const float* __restrict__ X,
                                               const float* __restrict__ W1,
                                               const float* __restrict__ b1,
                                               float* __restrict__ Y) {
    __shared__ __align__(32) __bf16 sW1t[TMP * HIDDEN]; // n-major: [n][k], 64KB
    int tid = threadIdx.x;
    for (int i = tid; i < TMP * HIDDEN; i += 256) {
        int n = i / HIDDEN, k = i % HIDDEN;
        sW1t[i] = (__bf16)W1[k * TMP + n];
    }
    __syncthreads();

    int wave = tid >> 5, lane = tid & 31;
    int m_ = lane & 15, h_ = lane >> 4;
    int t0 = blockIdx.x * 128;
    int trow = t0 + wave * 16 + m_;

    // A fragments: 8 k-blocks of K=32
    v16bf afrag[8];
#pragma unroll
    for (int kb = 0; kb < 8; ++kb) {
        int kbase = kb * 32;
#pragma unroll
        for (int i = 0; i < 8; ++i) {
            afrag[kb][i]     = (__bf16)X[(size_t)trow * HIDDEN + kbase + h_ * 16 + i];
            afrag[kb][8 + i] = (__bf16)X[(size_t)trow * HIDDEN + kbase + h_ * 16 + 8 + i];
        }
    }
#pragma unroll
    for (int nt = 0; nt < 8; ++nt) {
        int n = nt * 16 + (lane & 15);
        float bv = b1[n];
        v8f c;
#pragma unroll
        for (int r = 0; r < 8; ++r) c[r] = bv;
#pragma unroll
        for (int kb = 0; kb < 8; ++kb) {
            v16bf bfrag = *(const v16bf*)&sW1t[n * HIDDEN + kb * 32 + h_ * 16];
            c = __builtin_amdgcn_wmma_f32_16x16x32_bf16(false, afrag[kb], false, bfrag,
                                                        (short)0, c, false, false);
        }
#pragma unroll
        for (int r = 0; r < 8; ++r) {
            int row = t0 + wave * 16 + r + 8 * h_;
            Y[(size_t)row * TMP + n] = c[r];
        }
    }
}

// ---------------- Kernel 2: transition table (the big WMMA kernel) ----------------
// per t: Z = leaky(Y[t] + E) @ W2  (128x128x128, bf16) ; ARG[t][j] = argmax_n Z[j][n]
#define TB 16
__global__ __launch_bounds__(256) void k_table(const float* __restrict__ Y,
                                               const float* __restrict__ E,
                                               const float* __restrict__ W2,
                                               const float* __restrict__ b2,
                                               unsigned char* __restrict__ ARG) {
    __shared__ __align__(32) __bf16 sW2t[EVD * TMP];   // n-major [n][k], 32KB
    __shared__ __align__(32) __bf16 sE[128 * TMP];     // [j][k],          32KB
    __shared__ float sY[TB * TMP];                     // 8KB
    __shared__ float sb2[EVD];

    int tid = threadIdx.x;
    for (int i = tid; i < EVD * TMP; i += 256) {
        int n = i / TMP, k = i % TMP;
        sW2t[i] = (__bf16)W2[k * EVD + n];
        sE[i]   = (__bf16)E[i];
    }
    int t0 = blockIdx.x * TB;
    for (int i = tid; i < TB * TMP; i += 256) sY[i] = Y[(size_t)t0 * TMP + i];
    if (tid < EVD) sb2[tid] = b2[tid];
    __syncthreads();

    int wave = tid >> 5, lane = tid & 31;
    int m_ = lane & 15, h_ = lane >> 4;
    int jrow = wave * 16 + m_; // state index for A rows

    for (int tt = 0; tt < TB; ++tt) {
        const float* y = &sY[tt * TMP];
        v16bf afrag[4];
#pragma unroll
        for (int kb = 0; kb < 4; ++kb) {
            int kbase = kb * 32;
#pragma unroll
            for (int i = 0; i < 8; ++i) {
                int k1 = kbase + h_ * 16 + i;
                int k2 = kbase + h_ * 16 + 8 + i;
                float x1 = lrelu(y[k1] + (float)sE[jrow * TMP + k1]);
                float x2 = lrelu(y[k2] + (float)sE[jrow * TMP + k2]);
                afrag[kb][i]     = (__bf16)x1;
                afrag[kb][8 + i] = (__bf16)x2;
            }
        }
        float bestv[8];
        int   besti[8];
#pragma unroll
        for (int r = 0; r < 8; ++r) { bestv[r] = -3.4e38f; besti[r] = 0; }

#pragma unroll
        for (int nt = 0; nt < 8; ++nt) {
            int n = nt * 16 + (lane & 15);
            float bv = sb2[n];
            v8f c;
#pragma unroll
            for (int r = 0; r < 8; ++r) c[r] = bv;
#pragma unroll
            for (int kb = 0; kb < 4; ++kb) {
                v16bf bfrag = *(const v16bf*)&sW2t[n * TMP + kb * 32 + h_ * 16];
                c = __builtin_amdgcn_wmma_f32_16x16x32_bf16(false, afrag[kb], false, bfrag,
                                                            (short)0, c, false, false);
            }
#pragma unroll
            for (int r = 0; r < 8; ++r) {
                float v = c[r];
                if (v > bestv[r]) { bestv[r] = v; besti[r] = n; }
            }
        }
        // reduce across the 16 lanes of each half (C layout: N = lane%16)
#pragma unroll
        for (int r = 0; r < 8; ++r) {
            float v = bestv[r]; int ix = besti[r];
#pragma unroll
            for (int mk = 8; mk >= 1; mk >>= 1) {
                float ov = __shfl_xor(v, mk, 32);
                int   oi = __shfl_xor(ix, mk, 32);
                if (ov > v) { v = ov; ix = oi; }
            }
            if ((lane & 15) == 0) {
                int row = wave * 16 + 8 * h_ + r; // M = r + 8*(lane/16)
                ARG[(size_t)(t0 + tt) * 128 + row] = (unsigned char)ix;
            }
        }
    }
}

// ---------------- Kernel 3a: per-chunk FSM map composition ----------------
#define CH 512
#define NCH 128
__global__ __launch_bounds__(128) void k_fsm_chunk(const unsigned char* __restrict__ ARG,
                                                   unsigned char* __restrict__ CM) {
    __shared__ unsigned char sA[CH * 128]; // 64KB
    int tid = threadIdx.x, blk = blockIdx.x;
    const unsigned int* g = (const unsigned int*)(ARG + (size_t)blk * CH * 128);
    unsigned int* s32 = (unsigned int*)sA;
    for (int i = tid; i < CH * 128 / 4; i += 128) s32[i] = g[i];
    __syncthreads();
    int c = tid;
    for (int i = 0; i < CH; ++i) c = sA[i * 128 + c];
    CM[blk * 128 + tid] = (unsigned char)c;
}

// ---------------- Kernel 3b: serial chunk-start states (tiny) ----------------
__global__ void k_fsm_seq(const unsigned char* __restrict__ CM,
                          const int* __restrict__ evt,
                          int* __restrict__ start) {
    int s = evt[0];
    start[0] = s;
    for (int c = 0; c < NCH - 1; ++c) {
        s = CM[c * 128 + s];
        start[c + 1] = s;
    }
}

// ---------------- Kernel 3c: re-walk chunks, emit prev/next state per t ----------------
__global__ __launch_bounds__(128) void k_fsm_walk(const unsigned char* __restrict__ ARG,
                                                  const int* __restrict__ start,
                                                  int* __restrict__ pidx,
                                                  int* __restrict__ nidx) {
    __shared__ unsigned char sA[CH * 128];
    int tid = threadIdx.x, blk = blockIdx.x;
    const unsigned int* g = (const unsigned int*)(ARG + (size_t)blk * CH * 128);
    unsigned int* s32 = (unsigned int*)sA;
    for (int i = tid; i < CH * 128 / 4; i += 128) s32[i] = g[i];
    __syncthreads();
    if (tid == 0) {
        int c = start[blk];
        for (int i = 0; i < CH; ++i) {
            int t = blk * CH + i;
            pidx[t] = c;
            c = sA[i * 128 + c];
            nidx[t] = c;
        }
    }
}

// ---------------- Kernel 4: final log_softmax + outputs ----------------
// block = 256 (8 waves), each wave handles 8 timesteps
__global__ __launch_bounds__(256) void k_final(const float* __restrict__ Y,
                                               const float* __restrict__ E,
                                               const float* __restrict__ W2,
                                               const float* __restrict__ b2,
                                               const float* __restrict__ emb,
                                               const int* __restrict__ pidx,
                                               const int* __restrict__ nidx,
                                               float* __restrict__ out) {
    __shared__ float sW2[TMP * EVD]; // 64KB, [k][n]
    __shared__ float sH[8 * TMP];    // per-wave h
    int tid = threadIdx.x;
    for (int i = tid; i < TMP * EVD; i += 256) sW2[i] = W2[i];
    __syncthreads();

    int wave = tid >> 5, lane = tid & 31;
    int base = blockIdx.x * 64;

    for (int it = 0; it < 8; ++it) {
        int t = base + it * 8 + wave;
        int j = pidx[t];
        int ni = nidx[t];
        // h = leaky(Y[t] + E[j])
#pragma unroll
        for (int q = 0; q < 4; ++q) {
            int k = lane + q * 32;
            sH[wave * TMP + k] = lrelu(Y[(size_t)t * TMP + k] + E[j * TMP + k]);
        }
        __syncthreads();
        float acc[4];
#pragma unroll
        for (int q = 0; q < 4; ++q) acc[q] = b2[lane + q * 32];
        for (int k = 0; k < TMP; ++k) {
            float hk = sH[wave * TMP + k];
#pragma unroll
            for (int q = 0; q < 4; ++q)
                acc[q] = fmaf(hk, sW2[k * EVD + lane + q * 32], acc[q]);
        }
        // log_softmax across 128 logits (4 per lane)
        float m = fmaxf(fmaxf(acc[0], acc[1]), fmaxf(acc[2], acc[3]));
#pragma unroll
        for (int off = 16; off >= 1; off >>= 1) m = fmaxf(m, __shfl_xor(m, off, 32));
        float s = 0.0f;
#pragma unroll
        for (int q = 0; q < 4; ++q) s += expf(acc[q] - m);
#pragma unroll
        for (int off = 16; off >= 1; off >>= 1) s += __shfl_xor(s, off, 32);
        float lse = m + logf(s);
#pragma unroll
        for (int q = 0; q < 4; ++q)
            out[(size_t)t * EVD + lane + q * 32] = acc[q] - lse;
        if (lane < 16)
            out[(size_t)T_STEPS * EVD + (size_t)t * EMB + lane] = emb[ni * EMB + lane];
        if (lane == 0)
            out[(size_t)T_STEPS * EVD + (size_t)T_STEPS * EMB + t] = (float)ni;
        __syncthreads();
    }
}

extern "C" void kernel_launch(void* const* d_in, const int* in_sizes, int n_in,
                              void* d_out, int out_size, void* d_ws, size_t ws_size,
                              hipStream_t stream) {
    const float* X   = (const float*)d_in[0];
    const int*   evt = (const int*)d_in[1];
    const float* emb = (const float*)d_in[2];
    const float* W1  = (const float*)d_in[3];
    const float* b1  = (const float*)d_in[4];
    const float* W2  = (const float*)d_in[5];
    const float* b2  = (const float*)d_in[6];
    float* out = (float*)d_out;

    char* ws = (char*)d_ws;
    float*         Y     = (float*)(ws + WS_Y);
    unsigned char* ARG   = (unsigned char*)(ws + WS_ARG);
    float*         E     = (float*)(ws + WS_E);
    unsigned char* CM    = (unsigned char*)(ws + WS_CM);
    int*           START = (int*)(ws + WS_START);
    int*           PIDX  = (int*)(ws + WS_PIDX);
    int*           NIDX  = (int*)(ws + WS_NIDX);

    k_prepE<<<64, 256, 0, stream>>>(emb, W1, E);
    k_gemm1<<<T_STEPS / 128, 256, 0, stream>>>(X, W1, b1, Y);
    k_table<<<T_STEPS / TB, 256, 0, stream>>>(Y, E, W2, b2, ARG);
    k_fsm_chunk<<<NCH, 128, 0, stream>>>(ARG, CM);
    k_fsm_seq<<<1, 1, 0, stream>>>(CM, evt, START);
    k_fsm_walk<<<NCH, 128, 0, stream>>>(ARG, START, PIDX, NIDX);
    k_final<<<T_STEPS / 64, 256, 0, stream>>>(Y, E, W2, b2, emb, PIDX, NIDX, out);
}
